// Seq2Seq_85856396247046
// MI455X (gfx1250) — matrix operand order repaired and checked by reading, not compiled, not run on hardware
//
#include <hip/hip_runtime.h>
#include <hip/hip_bf16.h>
#include <math.h>

// Problem constants (from reference)
constexpr int V = 32000, E = 256, H = 512, B = 32, S = 128, T = 64;
constexpr int H2 = 2 * H;        // 1024
constexpr int H3 = 3 * H;        // 1536
constexpr int KD = E + H2;       // 1280  decoder GRU input width
constexpr int KF = H3 + E;       // 1792  fc1 input width  (wait: 3H+E) -> 1536+256=1792

typedef __attribute__((ext_vector_type(16))) __bf16 v16bf;
typedef __attribute__((ext_vector_type(8)))  __bf16 v8bf;
typedef __attribute__((ext_vector_type(8)))  float  v8f;

// ---------------------------------------------------------------------------
// Generic WMMA GEMM:  C[M,N] = A[M,K](bf16,row-major) * W[N,K]^T(bf16) + bias
// One wave per 16x16 C tile, 8 waves per block. K stepped by 32 using
// v_wmma_f32_16x16x32_bf16. Fragment layouts per CDNA5 ISA 7.12.2 (wave32):
//   A 16x32: lanes 0-15 -> M=lane, K groups {0..7,16..23}; lanes 16-31 -> +8
//   B 32x16: lanes 0-15 -> K {0..7,16..23}; lanes 16-31 -> +8 ; N = lane&15
//   C/D:     lane<16 -> M = r ; lane>=16 -> M = 8+r ; N = lane&15
// ---------------------------------------------------------------------------
__global__ __launch_bounds__(256)
void k_gemm(const __bf16* __restrict__ A, const __bf16* __restrict__ W,
            const float* __restrict__ bias, float* __restrict__ C,
            int M, int N, int K, long long ldc)
{
  const int lane = threadIdx.x & 31;
  const int wave = threadIdx.x >> 5;
  const int tn = blockIdx.x * 8 + wave;   // N-tile
  const int tm = blockIdx.y;              // M-tile
  if (tn * 16 >= N) return;

  const int hi   = lane >> 4;             // 0: lanes 0-15, 1: lanes 16-31
  const int mrow = tm * 16 + (lane & 15);
  const int ncol = tn * 16 + (lane & 15);

  const __bf16* arow = A + (size_t)mrow * K + (hi ? 8 : 0);
  const __bf16* wrow = W + (size_t)ncol * K + (hi ? 8 : 0);

  v8f acc = {};
  for (int k0 = 0; k0 < K; k0 += 32) {
    v8bf a0 = *(const v8bf*)(arow + k0);
    v8bf a1 = *(const v8bf*)(arow + k0 + 16);
    v8bf b0 = *(const v8bf*)(wrow + k0);
    v8bf b1 = *(const v8bf*)(wrow + k0 + 16);
    v16bf av, bv;
#pragma unroll
    for (int i = 0; i < 8; ++i) {
      av[i] = a0[i]; av[8 + i] = a1[i];
      bv[i] = b0[i]; bv[8 + i] = b1[i];
    }
    // keep the weight stream (L2-resident bf16) ahead of the WMMA pipe
    __builtin_prefetch((const void*)(wrow + k0 + 256), 0, 0);
    acc = __builtin_amdgcn_wmma_f32_16x16x32_bf16(
        /*neg_a=*/false, av, /*neg_b=*/false, bv,
        /*c_mod=*/(short)0, acc, /*reuse_a=*/false, /*reuse_b=*/false);
  }

  const float badd = bias ? bias[ncol] : 0.0f;
  const int rbase = tm * 16 + (hi ? 8 : 0);
#pragma unroll
  for (int r = 0; r < 8; ++r)
    C[(long long)(rbase + r) * ldc + ncol] = acc[r] + badd;
}

// ---------------------------------------------------------------------------
// fp32 -> bf16 strided cast (weight conversion / packing)
// ---------------------------------------------------------------------------
__global__ void k_cast(const float* __restrict__ in, long long in_stride,
                       __bf16* __restrict__ out, long long out_stride,
                       int rows, int cols)
{
  long long idx = (long long)blockIdx.x * 256 + threadIdx.x;
  if (idx >= (long long)rows * cols) return;
  int r = (int)(idx / cols), c = (int)(idx % cols);
  out[(long long)r * out_stride + c] = (__bf16)in[(long long)r * in_stride + c];
}

// encoder embedding gather: X[(s*B+b), e] = enc_emb[src[b,s], e]   (bf16)
__global__ void k_embed_enc(const int* __restrict__ src,
                            const float* __restrict__ emb,
                            __bf16* __restrict__ X)
{
  int idx = blockIdx.x * 256 + threadIdx.x;   // S*B*E
  if (idx >= S * B * E) return;
  int e = idx % E;
  int sb = idx / E;
  int b = sb % B, s = sb / B;
  int tok = src[b * S + s];
  X[idx] = (__bf16)emb[(size_t)tok * E + e];
}

// decoder embedding gather for step t: fills gin[:,0:E] and z[:,H:H+E]
__global__ void k_dec_emb(const int* __restrict__ target, int t,
                          const float* __restrict__ emb,
                          __bf16* __restrict__ gin, __bf16* __restrict__ z)
{
  int idx = blockIdx.x * 256 + threadIdx.x;   // B*E
  if (idx >= B * E) return;
  int b = idx / E, e = idx % E;
  int tok = target[b * T + t];
  float v = emb[(size_t)tok * E + e];
  gin[(size_t)b * KD + e]     = (__bf16)v;
  z[(size_t)b * KF + H + e]   = (__bf16)v;
}

// GRU elementwise: r=sig(xr+hr) z=sig(xz+hz) n=tanh(xn+r*hn) h'=(1-z)n+z h
__global__ void k_gru_elem(const float* __restrict__ gx, const float* __restrict__ gh,
                           const float* __restrict__ h_in,
                           float* __restrict__ h_out, __bf16* __restrict__ h_out_bf,
                           float* __restrict__ eo, long long eo_stride,
                           __bf16* __restrict__ eo_bf, long long eob_stride)
{
  int idx = blockIdx.x * 256 + threadIdx.x;   // B*H
  if (idx >= B * H) return;
  int b = idx / H, j = idx % H;
  const float* gxr = gx + (size_t)b * H3;
  const float* ghr = gh + (size_t)b * H3;
  float r = 1.0f / (1.0f + expf(-(gxr[j]          + ghr[j])));
  float z = 1.0f / (1.0f + expf(-(gxr[H + j]      + ghr[H + j])));
  float n = tanhf(gxr[2 * H + j] + r * ghr[2 * H + j]);
  float hn = (1.0f - z) * n + z * h_in[idx];
  h_out[idx]    = hn;
  h_out_bf[idx] = (__bf16)hn;
  if (eo)    eo[(long long)b * eo_stride + j]      = hn;
  if (eo_bf) eo_bf[(long long)b * eob_stride + j]  = (__bf16)hn;
}

__global__ void k_zero_state(float* a, __bf16* bf, int n)
{
  int idx = blockIdx.x * 256 + threadIdx.x;
  if (idx >= n) return;
  a[idx] = 0.0f;
  bf[idx] = (__bf16)0.0f;
}

__global__ void k_tanh_cast(const float* __restrict__ in,
                            float* __restrict__ out_f, __bf16* __restrict__ out_bf, int n)
{
  int idx = blockIdx.x * 256 + threadIdx.x;
  if (idx >= n) return;
  float v = tanhf(in[idx]);
  out_f[idx]  = v;
  out_bf[idx] = (__bf16)v;
}

// scores[b,s] = sum_h tanh(hW[b,h] + proj[b,s,h] + ab[h]) * av[h]  (masked)
__global__ void k_attn_scores(const float* __restrict__ hW, const float* __restrict__ proj,
                              const float* __restrict__ ab, const float* __restrict__ av,
                              const int* __restrict__ src, float* __restrict__ scores)
{
  int idx = blockIdx.x * 256 + threadIdx.x;   // B*S
  if (idx >= B * S) return;
  int b = idx / S, s = idx % S;
  const float* pr = proj + (size_t)idx * H;
  const float* hw = hW + (size_t)b * H;
  float acc = 0.0f;
  for (int h = 0; h < H; ++h)
    acc += tanhf(hw[h] + pr[h] + ab[h]) * av[h];
  scores[idx] = (src[b * S + s] != 0) ? acc : -1.0e10f;
}

// per-batch masked softmax over S + weighted sum over enc_bt -> gin/z (bf16)
__global__ void k_softmax_weighted(const float* __restrict__ scores,
                                   const float* __restrict__ encbt,
                                   __bf16* __restrict__ gin, __bf16* __restrict__ z)
{
  int b = blockIdx.x, tid = threadIdx.x;
  __shared__ float sa[S];
  __shared__ float smax, ssum;
  if (tid < S) sa[tid] = scores[b * S + tid];
  __syncthreads();
  if (tid == 0) {
    float m = sa[0];
    for (int i = 1; i < S; ++i) m = fmaxf(m, sa[i]);
    smax = m;
  }
  __syncthreads();
  if (tid < S) sa[tid] = expf(sa[tid] - smax);
  __syncthreads();
  if (tid == 0) {
    float sum = 0.0f;
    for (int i = 0; i < S; ++i) sum += sa[i];
    ssum = sum;
  }
  __syncthreads();
  float inv = 1.0f / ssum;
  for (int d = tid; d < H2; d += blockDim.x) {
    float acc = 0.0f;
    const float* col = encbt + (size_t)b * S * H2 + d;
    for (int s = 0; s < S; ++s) acc += sa[s] * col[(size_t)s * H2];
    __bf16 bv = (__bf16)(acc * inv);
    gin[(size_t)b * KD + E + d]     = bv;
    z[(size_t)b * KF + H + E + d]   = bv;
  }
}

// per-batch argmax over V with first-index tie-break; preds as float
__global__ void k_argmax(const float* __restrict__ out, float* __restrict__ preds, int t)
{
  int b = blockIdx.x, tid = threadIdx.x;
  const float* row = out + (size_t)b * T * V + (size_t)t * V;
  float best = -INFINITY; int bi = 0x7fffffff;
  for (int v = tid; v < V; v += 256) {
    float x = row[v];
    if (x > best || (x == best && v < bi)) { best = x; bi = v; }
  }
  __shared__ float sv[256];
  __shared__ int   si[256];
  sv[tid] = best; si[tid] = bi;
  __syncthreads();
  for (int off = 128; off > 0; off >>= 1) {
    if (tid < off) {
      if (sv[tid + off] > sv[tid] ||
          (sv[tid + off] == sv[tid] && si[tid + off] < si[tid])) {
        sv[tid] = sv[tid + off]; si[tid] = si[tid + off];
      }
    }
    __syncthreads();
  }
  if (tid == 0) preds[b * T + t] = (float)si[0];
}

// zero outputs[:, T-1, :] and preds[:, T-1]
__global__ void k_zero_tail(float* __restrict__ out)
{
  int idx = blockIdx.x * 256 + threadIdx.x;
  if (idx < B * V) {
    int b = idx / V, v = idx % V;
    out[(size_t)b * T * V + (size_t)(T - 1) * V + v] = 0.0f;
  }
  if (idx < B)
    out[(size_t)B * T * V + (size_t)idx * T + (T - 1)] = 0.0f;
}

// ---------------------------------------------------------------------------
extern "C" void kernel_launch(void* const* d_in, const int* in_sizes, int n_in,
                              void* d_out, int out_size, void* d_ws, size_t ws_size,
                              hipStream_t stream)
{
  (void)in_sizes; (void)n_in; (void)out_size; (void)ws_size;

  const int*   src       = (const int*)d_in[0];
  const int*   target    = (const int*)d_in[1];
  const float* enc_emb   = (const float*)d_in[2];
  const float* enc_Wih_f = (const float*)d_in[3];
  const float* enc_Whh_f = (const float*)d_in[4];
  const float* enc_bih_f = (const float*)d_in[5];
  const float* enc_bhh_f = (const float*)d_in[6];
  const float* enc_Wih_b = (const float*)d_in[7];
  const float* enc_Whh_b = (const float*)d_in[8];
  const float* enc_bih_b = (const float*)d_in[9];
  const float* enc_bhh_b = (const float*)d_in[10];
  const float* enc_fc_W  = (const float*)d_in[11];
  const float* enc_fc_b  = (const float*)d_in[12];
  const float* dec_emb   = (const float*)d_in[13];
  const float* attn_W    = (const float*)d_in[14];
  const float* attn_b    = (const float*)d_in[15];
  const float* attn_v    = (const float*)d_in[16];
  const float* dec_Wih   = (const float*)d_in[17];
  const float* dec_Whh   = (const float*)d_in[18];
  const float* dec_bih   = (const float*)d_in[19];
  const float* dec_bhh   = (const float*)d_in[20];
  const float* fc1_W     = (const float*)d_in[21];
  const float* fc1_b     = (const float*)d_in[22];

  float* outp  = (float*)d_out;                       // [B,T,V]
  float* preds = (float*)d_out + (size_t)B * T * V;   // [B,T]

  // ---- workspace bump allocator --------------------------------------------
  char* ws = (char*)d_ws;
  size_t off = 0;
  auto alloc = [&](size_t bytes) -> void* {
    void* p = ws + off;
    off = (off + bytes + 255) & ~(size_t)255;
    return p;
  };
  // bf16 weights (L2-resident working set; fc1 = 114.7 MB fits 192 MB L2)
  __bf16* wih_f  = (__bf16*)alloc((size_t)H3 * E * 2);
  __bf16* whh_f  = (__bf16*)alloc((size_t)H3 * H * 2);
  __bf16* wih_b  = (__bf16*)alloc((size_t)H3 * E * 2);
  __bf16* whh_b  = (__bf16*)alloc((size_t)H3 * H * 2);
  __bf16* fcW    = (__bf16*)alloc((size_t)H * H2 * 2);
  __bf16* Wh     = (__bf16*)alloc((size_t)H * H * 2);
  __bf16* We     = (__bf16*)alloc((size_t)H * H2 * 2);
  __bf16* dWih   = (__bf16*)alloc((size_t)H3 * KD * 2);
  __bf16* dWhh   = (__bf16*)alloc((size_t)H3 * H * 2);
  __bf16* f1W    = (__bf16*)alloc((size_t)V * KF * 2);
  // bf16 activations
  __bf16* X      = (__bf16*)alloc((size_t)S * B * E * 2);
  __bf16* ebt_bf = (__bf16*)alloc((size_t)B * S * H2 * 2);
  __bf16* hf_bf  = (__bf16*)alloc((size_t)B * H * 2);
  __bf16* hb_bf  = (__bf16*)alloc((size_t)B * H * 2);
  __bf16* hcat   = (__bf16*)alloc((size_t)B * H2 * 2);
  __bf16* hd_bf  = (__bf16*)alloc((size_t)B * H * 2);
  __bf16* gin_bf = (__bf16*)alloc((size_t)B * KD * 2);
  __bf16* z_bf   = (__bf16*)alloc((size_t)B * KF * 2);
  // f32 activations
  float* gx_f    = (float*)alloc((size_t)S * B * H3 * 4);
  float* gx_b    = (float*)alloc((size_t)S * B * H3 * 4);
  float* ebt_f   = (float*)alloc((size_t)B * S * H2 * 4);
  float* hf_f    = (float*)alloc((size_t)B * H * 4);
  float* hb_f    = (float*)alloc((size_t)B * H * 4);
  float* gh_buf  = (float*)alloc((size_t)B * H3 * 4);
  float* gxd_buf = (float*)alloc((size_t)B * H3 * 4);
  float* hd_f    = (float*)alloc((size_t)B * H * 4);
  float* hid_pre = (float*)alloc((size_t)B * H * 4);
  float* eproj   = (float*)alloc((size_t)B * S * H * 4);
  float* hW      = (float*)alloc((size_t)B * H * 4);
  float* scores  = (float*)alloc((size_t)B * S * 4);

  auto cdiv = [](long long a, long long b) { return (int)((a + b - 1) / b); };
  auto cast = [&](const float* in, long long istr, __bf16* out, long long ostr,
                  int rows, int cols) {
    k_cast<<<cdiv((long long)rows * cols, 256), 256, 0, stream>>>(in, istr, out, ostr, rows, cols);
  };

  // ---- phase 0: weight conversion to bf16 ---------------------------------
  cast(enc_Wih_f, E,  wih_f, E,  H3, E);
  cast(enc_Whh_f, H,  whh_f, H,  H3, H);
  cast(enc_Wih_b, E,  wih_b, E,  H3, E);
  cast(enc_Whh_b, H,  whh_b, H,  H3, H);
  cast(enc_fc_W,  H2, fcW,   H2, H,  H2);
  cast(attn_W,      H3, Wh, H,  H, H);        // attn_W[:, :H]
  cast(attn_W + H,  H3, We, H2, H, H2);       // attn_W[:, H:]
  cast(dec_Wih,   KD, dWih,  KD, H3, KD);
  cast(dec_Whh,   H,  dWhh,  H,  H3, H);
  cast(fc1_W,     KF, f1W,   KF, V,  KF);

  // ---- phase 1: encoder embeddings + batched input-side GEMMs -------------
  k_embed_enc<<<cdiv((long long)S * B * E, 256), 256, 0, stream>>>(src, enc_emb, X);
  // gx[s,b,:] = X @ Wih^T + bih   (M=4096, N=1536, K=256)
  k_gemm<<<dim3(H3 / 16 / 8, (S * B) / 16), 256, 0, stream>>>(
      X, wih_f, enc_bih_f, gx_f, S * B, H3, E, H3);
  k_gemm<<<dim3(H3 / 16 / 8, (S * B) / 16), 256, 0, stream>>>(
      X, wih_b, enc_bih_b, gx_b, S * B, H3, E, H3);

  // ---- phase 2: encoder recurrent scans -----------------------------------
  k_zero_state<<<cdiv(B * H, 256), 256, 0, stream>>>(hf_f, hf_bf, B * H);
  k_zero_state<<<cdiv(B * H, 256), 256, 0, stream>>>(hb_f, hb_bf, B * H);

  for (int s = 0; s < S; ++s) {  // forward
    k_gemm<<<dim3(H3 / 16 / 8, B / 16), 256, 0, stream>>>(
        hf_bf, whh_f, enc_bhh_f, gh_buf, B, H3, H, H3);
    k_gru_elem<<<cdiv(B * H, 256), 256, 0, stream>>>(
        gx_f + (size_t)s * B * H3, gh_buf, hf_f, hf_f, hf_bf,
        ebt_f + (size_t)s * H2, (long long)S * H2,
        ebt_bf + (size_t)s * H2, (long long)S * H2);
  }
  for (int i = 0; i < S; ++i) {  // backward (process s = S-1 .. 0)
    int s = S - 1 - i;
    k_gemm<<<dim3(H3 / 16 / 8, B / 16), 256, 0, stream>>>(
        hb_bf, whh_b, enc_bhh_b, gh_buf, B, H3, H, H3);
    k_gru_elem<<<cdiv(B * H, 256), 256, 0, stream>>>(
        gx_b + (size_t)s * B * H3, gh_buf, hb_f, hb_f, hb_bf,
        ebt_f + (size_t)s * H2 + H, (long long)S * H2,
        ebt_bf + (size_t)s * H2 + H, (long long)S * H2);
  }

  // ---- phase 3: decoder init hidden = tanh([hf,hb] @ fcW^T + b) -----------
  cast(hf_f, H, hcat,     H2, B, H);
  cast(hb_f, H, hcat + H, H2, B, H);
  k_gemm<<<dim3(H / 16 / 8, B / 16), 256, 0, stream>>>(
      hcat, fcW, enc_fc_b, hid_pre, B, H, H2, H);
  k_tanh_cast<<<cdiv(B * H, 256), 256, 0, stream>>>(hid_pre, hd_f, hd_bf, B * H);

  // ---- phase 4: enc_proj = enc_bt @ W_e^T  (M=4096, N=512, K=1024) --------
  k_gemm<<<dim3(H / 16 / 8, (B * S) / 16), 256, 0, stream>>>(
      ebt_bf, We, nullptr, eproj, B * S, H, H2, H);

  // ---- phase 5: decoder steps ---------------------------------------------
  for (int t = 0; t < T - 1; ++t) {
    k_dec_emb<<<cdiv(B * E, 256), 256, 0, stream>>>(target, t, dec_emb, gin_bf, z_bf);
    // hW = h @ W_h^T
    k_gemm<<<dim3(H / 16 / 8, B / 16), 256, 0, stream>>>(
        hd_bf, Wh, nullptr, hW, B, H, H, H);
    k_attn_scores<<<cdiv(B * S, 256), 256, 0, stream>>>(hW, eproj, attn_b, attn_v, src, scores);
    k_softmax_weighted<<<B, 256, 0, stream>>>(scores, ebt_f, gin_bf, z_bf);
    // GRU: gx = [emb,weighted] @ dWih^T + bih ;  gh = h @ dWhh^T + bhh
    k_gemm<<<dim3(H3 / 16 / 8, B / 16), 256, 0, stream>>>(
        gin_bf, dWih, dec_bih, gxd_buf, B, H3, KD, H3);
    k_gemm<<<dim3(H3 / 16 / 8, B / 16), 256, 0, stream>>>(
        hd_bf, dWhh, dec_bhh, gh_buf, B, H3, H, H3);
    // elementwise update; also drop h2 (bf16) into z[:, 0:H]
    k_gru_elem<<<cdiv(B * H, 256), 256, 0, stream>>>(
        gxd_buf, gh_buf, hd_f, hd_f, hd_bf,
        nullptr, 0, z_bf, (long long)KF);
    // fc1: out[:, t, :] = [h2, emb, weighted] @ fc1_W^T + b   (dominant GEMM)
    k_gemm<<<dim3(V / 16 / 8, B / 16), 256, 0, stream>>>(
        z_bf, f1W, fc1_b, outp + (size_t)t * V, B, V, KF, (long long)T * V);
    k_argmax<<<B, 256, 0, stream>>>(outp, preds, t);
  }

  // ---- phase 6: zero last timestep (reference leaves t=T-1 at zero) -------
  k_zero_tail<<<cdiv(B * V, 256), 256, 0, stream>>>(outp);
}
